// APLKuramotoLayer_15925738734066
// MI455X (gfx1250) — matrix operand order repaired and checked by reading, not compiled, not run on hardware
//
#include <hip/hip_runtime.h>

#define N 1024
#define BATCH 128
#define DT 0.1f
#define STEPS 10

typedef __attribute__((ext_vector_type(16))) _Float16 v16h;
typedef __attribute__((ext_vector_type(8)))  _Float16 v8h;
typedef __attribute__((ext_vector_type(8)))  float    v8f;

// ---------------------------------------------------------------------------
// Convert coupling matrix K (f32, symmetric) to f16 once per launch.
// ---------------------------------------------------------------------------
__global__ void kuramoto_convert_k(const float* __restrict__ K,
                                   _Float16* __restrict__ Kh) {
    int idx = blockIdx.x * blockDim.x + threadIdx.x;
    Kh[idx] = (_Float16)K[idx];
}

// ---------------------------------------------------------------------------
// Initialize state: copy theta into workspace, emit f16 sin/cos operands.
// ---------------------------------------------------------------------------
__global__ void kuramoto_init(const float* __restrict__ theta_init,
                              float* __restrict__ theta,
                              _Float16* __restrict__ sin_h,
                              _Float16* __restrict__ cos_h) {
    int idx = blockIdx.x * blockDim.x + threadIdx.x;
    float t = theta_init[idx];
    theta[idx] = t;
    sin_h[idx] = (_Float16)sinf(t);
    cos_h[idx] = (_Float16)cosf(t);
}

// ---------------------------------------------------------------------------
// Fused Kuramoto step:
//   accS = SinTheta @ K, accC = CosTheta @ K   (K symmetric -> B-operand
//   "columns" are contiguous rows), then per-tile Euler update + wrap,
//   emitting next step's f16 sin/cos into the ping-pong buffers.
//
// Grid: N/16 blocks (one 16-column tile), 256 threads = 8 waves; wave w owns
// batch-row tile w (8*16 = 128 = BATCH). The 16 K rows this block needs are
// staged once into LDS (32 KB) and shared by all 8 waves; sin/cos A-streams
// come straight from global (L2-resident, 256 KB each).
// ---------------------------------------------------------------------------
__global__ void __launch_bounds__(256)
kuramoto_step_fused(const _Float16* __restrict__ sin_in,
                    const _Float16* __restrict__ cos_in,
                    const _Float16* __restrict__ Kh,
                    const float* theta_in,          // may alias theta_out
                    const float* __restrict__ omega,
                    const float* __restrict__ Kg,
                    float* theta_out,
                    _Float16* __restrict__ sin_out,
                    _Float16* __restrict__ cos_out) {
    __shared__ _Float16 Btile[16 * N];  // 32 KB: K rows colbase..colbase+15

    const int tid    = threadIdx.x;
    const int lane   = tid & 31;
    const int wave   = tid >> 5;
    const int lane16 = lane & 15;
    const int lanehi = lane >> 4;            // 0: lanes 0-15, 1: lanes 16-31
    const int rowbase = wave * 16;           // batch-row tile
    const int colbase = blockIdx.x * 16;     // oscillator-column tile

    // Stage the block's 16 K rows into LDS: 256 threads x 8 x 16B = 32 KB.
    for (int off = tid * 8; off < 16 * N; off += 256 * 8) {
        int r    = off >> 10;                // off / N
        int cidx = off & (N - 1);            // off % N
        *(v8h*)(&Btile[off]) =
            *(const v8h*)(Kh + (size_t)(colbase + r) * N + cidx);
    }
    __syncthreads();

    // A-fragment per-lane addressing (ISA 7.12.2, 16-bit A 16x32):
    //   lane<16 : M=lane,    elems 0..7 -> K 0..7,   elems 8..15 -> K 16..23
    //   lane>=16: M=lane-16, elems 0..7 -> K 8..15,  elems 8..15 -> K 24..31
    const int aoff_lo = lanehi * 8;
    const int aoff_hi = 16 + lanehi * 8;

    const _Float16* arow_s = sin_in + (size_t)(rowbase + lane16) * N;
    const _Float16* arow_c = cos_in + (size_t)(rowbase + lane16) * N;
    // B-fragment (32x16 f16): elems 0..7 -> K-row lanehi*8+e, elems 8..15 ->
    // K-row 16+lanehi*8+(e-8), col lane16 (two 16B runs of staged row lane16).
    const _Float16* brow = &Btile[lane16 * N + lanehi * 8];

    v8f accS = {};
    v8f accC = {};

    for (int kbase = 0; kbase < N; kbase += 32) {
        if (kbase + 32 < N) {
            __builtin_prefetch(arow_s + kbase + 32, 0, 3);
            __builtin_prefetch(arow_c + kbase + 32, 0, 3);
        }

        v8h as_lo = *(const v8h*)(arow_s + kbase + aoff_lo);
        v8h as_hi = *(const v8h*)(arow_s + kbase + aoff_hi);
        v8h ac_lo = *(const v8h*)(arow_c + kbase + aoff_lo);
        v8h ac_hi = *(const v8h*)(arow_c + kbase + aoff_hi);
        v8h b_lo  = *(const v8h*)(brow + kbase);
        v8h b_hi  = *(const v8h*)(brow + kbase + 16);

        v16h a_s = __builtin_shufflevector(as_lo, as_hi,
                     0,1,2,3,4,5,6,7,8,9,10,11,12,13,14,15);
        v16h a_c = __builtin_shufflevector(ac_lo, ac_hi,
                     0,1,2,3,4,5,6,7,8,9,10,11,12,13,14,15);
        v16h b   = __builtin_shufflevector(b_lo, b_hi,
                     0,1,2,3,4,5,6,7,8,9,10,11,12,13,14,15);

        accS = __builtin_amdgcn_wmma_f32_16x16x32_f16(
                   false, a_s, false, b, (short)0, accS, false, false);
        accC = __builtin_amdgcn_wmma_f32_16x16x32_f16(
                   false, a_c, false, b, (short)0, accC, false, false);
    }

    // Fused Euler update on the D tile (ISA 7.12.2, 32-bit C/D 16x16):
    //   lane<16 -> col=lane, row=vgpr; lane>=16 -> col=lane-16, row=vgpr+8.
    const float kgn = Kg[0] * (1.0f / (float)N);
    const int   col = colbase + lane16;
    const float om  = omega[col];
#pragma unroll
    for (int r = 0; r < 8; ++r) {
        int row = rowbase + r + 8 * lanehi;
        size_t idx = (size_t)row * N + col;
        float t = theta_in[idx];
        float s = sinf(t);
        float c = cosf(t);
        float coupling = c * accS[r] - s * accC[r];
        float tn = t + DT * (om + kgn * coupling);
        float s2 = sinf(tn);
        float c2 = cosf(tn);
        theta_out[idx] = atan2f(s2, c2);  // wrap to (-pi, pi]
        sin_out[idx] = (_Float16)s2;      // sin/cos invariant under wrapping
        cos_out[idx] = (_Float16)c2;
    }
}

// ---------------------------------------------------------------------------
// Kuramoto order parameter r = |<e^{i theta}>| per batch row.
// ---------------------------------------------------------------------------
__global__ void kuramoto_coherence(const float* __restrict__ theta,
                                   float* __restrict__ coh) {
    const int b = blockIdx.x;
    const float* row = theta + (size_t)b * N;

    float cs = 0.0f, sn = 0.0f;
    for (int i = threadIdx.x; i < N; i += blockDim.x) {
        float t = row[i];
        cs += cosf(t);
        sn += sinf(t);
    }
    for (int off = 16; off > 0; off >>= 1) {
        cs += __shfl_down(cs, off, 32);
        sn += __shfl_down(sn, off, 32);
    }
    __shared__ float scs[8];
    __shared__ float ssn[8];
    int wave = threadIdx.x >> 5;
    int lane = threadIdx.x & 31;
    if (lane == 0) { scs[wave] = cs; ssn[wave] = sn; }
    __syncthreads();
    if (threadIdx.x == 0) {
        float C = 0.0f, S = 0.0f;
        for (int w = 0; w < 8; ++w) { C += scs[w]; S += ssn[w]; }
        C *= (1.0f / (float)N);
        S *= (1.0f / (float)N);
        coh[b] = sqrtf(C * C + S * S);
    }
}

// ---------------------------------------------------------------------------
// Launcher. Inputs: theta_init [B,N] f32, K [N,N] f32, omega [N] f32,
// K_global [1] f32. Output: theta [B,N] f32 ++ coherence [B] f32.
// ---------------------------------------------------------------------------
extern "C" void kernel_launch(void* const* d_in, const int* in_sizes, int n_in,
                              void* d_out, int out_size, void* d_ws, size_t ws_size,
                              hipStream_t stream) {
    const float* theta_init = (const float*)d_in[0];
    const float* K          = (const float*)d_in[1];
    const float* omega      = (const float*)d_in[2];
    const float* Kg         = (const float*)d_in[3];
    float* out = (float*)d_out;

    char* w = (char*)d_ws;
    _Float16* Kh   = (_Float16*)w;  w += (size_t)N * N * sizeof(_Float16);
    _Float16* sin0 = (_Float16*)w;  w += (size_t)BATCH * N * sizeof(_Float16);
    _Float16* cos0 = (_Float16*)w;  w += (size_t)BATCH * N * sizeof(_Float16);
    _Float16* sin1 = (_Float16*)w;  w += (size_t)BATCH * N * sizeof(_Float16);
    _Float16* cos1 = (_Float16*)w;  w += (size_t)BATCH * N * sizeof(_Float16);
    float*    theta = (float*)w;

    kuramoto_convert_k<<<(N * N) / 256, 256, 0, stream>>>(K, Kh);
    kuramoto_init<<<(BATCH * N) / 256, 256, 0, stream>>>(theta_init, theta,
                                                         sin0, cos0);

    for (int step = 0; step < STEPS; ++step) {
        const _Float16* si = (step & 1) ? sin1 : sin0;
        const _Float16* ci = (step & 1) ? cos1 : cos0;
        _Float16*       so = (step & 1) ? sin0 : sin1;
        _Float16*       co = (step & 1) ? cos0 : cos1;
        float* tdst = (step == STEPS - 1) ? out : theta;

        kuramoto_step_fused<<<N / 16, 256, 0, stream>>>(
            si, ci, Kh, theta, omega, Kg, tdst, so, co);
    }

    kuramoto_coherence<<<BATCH, 256, 0, stream>>>(out, out + (size_t)BATCH * N);
}
